// NucleusMoECrossAttention_45792941310318
// MI455X (gfx1250) — matrix operand order repaired and checked by reading, not compiled, not run on hardware
//
#include <hip/hip_runtime.h>

// ---------------------------------------------------------------------------
// MI455X (gfx1250) implementation of the joint cross-attention block.
// All GEMMs and attention matmuls run on v_wmma_f32_16x16x32_bf16 (wave32).
// Global->LDS staging uses GLOBAL_LOAD_ASYNC_TO_LDS_B128 (ASYNCcnt) with
// double-buffered ping-pong so copies overlap the WMMA phase.
// ---------------------------------------------------------------------------

typedef __attribute__((ext_vector_type(16))) __bf16 v16bf;
typedef __attribute__((ext_vector_type(8)))  __bf16 v8bf;
typedef __attribute__((ext_vector_type(8)))  float  v8f;

#define WMMA_BF16(a, b, c) \
  __builtin_amdgcn_wmma_f32_16x16x32_bf16(false, (a), false, (b), (short)0, (c), false, false)

// A-matrix fragment (16x32, 16-bit): lane = row (mod 16),
// K chunks {0..7,16..23} for lanes 0-15, {8..15,24..31} for lanes 16-31.
static __device__ inline v16bf frag_ld_a(const __bf16* tile, int ld) {
  const int lane = threadIdx.x & 31;
  const int row  = lane & 15;
  const int cb   = (lane >> 4) << 3;   // 0 or 8
  const __bf16* p = tile + (size_t)row * ld + cb;
  v8bf lo = *(const v8bf*)p;
  v8bf hi = *(const v8bf*)(p + 16);
  v16bf f;
#pragma unroll
  for (int i = 0; i < 8; ++i) { f[i] = lo[i]; f[i + 8] = hi[i]; }
  return f;
}

// B-matrix fragment (32x16, 16-bit): lane = column (mod 16),
// K chunks {0..7,16..23} for lanes 0-15, {8..15,24..31} for lanes 16-31.
static __device__ inline v16bf frag_ld_b(const __bf16* tile, int ld) {
  const int lane = threadIdx.x & 31;
  const int col  = lane & 15;
  const int kh   = (lane >> 4) << 3;   // 0 or 8
  const __bf16* p = tile + (size_t)col * ld + kh;
  v8bf lo = *(const v8bf*)p;
  v8bf hi = *(const v8bf*)(p + 16);
  v16bf f;
#pragma unroll
  for (int i = 0; i < 8; ++i) { f[i] = lo[i]; f[i + 8] = hi[i]; }
  return f;
}

// Async memory->LDS copy of 32 bytes (two b128), tracked by ASYNCcnt.
// INST_OFFSET is added to BOTH the LDS and global addresses (ISA 08 §4.4).
static __device__ inline void async_copy_32B(unsigned lds_addr, const __bf16* gaddr) {
  asm volatile(
      "global_load_async_to_lds_b128 %0, %1, off\n\t"
      "global_load_async_to_lds_b128 %0, %1, off offset:16"
      :: "v"(lds_addr), "v"(gaddr) : "memory");
}

static __device__ inline void async_wait0() {
  asm volatile("s_wait_asynccnt 0" ::: "memory");
}

// low 32 bits of a generic pointer to LDS == LDS byte address (flat aperture rule)
static __device__ inline unsigned lds_addr32(const void* p) {
  return (unsigned)(unsigned long long)p;
}

// ---------------------------------------------------------------------------
// f32 -> bf16 convert
// ---------------------------------------------------------------------------
__global__ __launch_bounds__(256) void cvt_f32_bf16_kernel(
    const float* __restrict__ in, __bf16* __restrict__ out, size_t n) {
  size_t i = (size_t)blockIdx.x * 256 + threadIdx.x;
  if (i < n) out[i] = (__bf16)in[i];
}

// f32 W[K][N] -> bf16 Wt[N][K]  (LDS 32x33 tile transpose, coalesced both ways)
__global__ __launch_bounds__(256) void transpose_cvt_kernel(
    const float* __restrict__ W, __bf16* __restrict__ Wt, int K, int N) {
  __shared__ float tile[32][33];
  const int n0 = blockIdx.x * 32;
  const int k0 = blockIdx.y * 32;
  const int tx = threadIdx.x & 31;
  const int ty = threadIdx.x >> 5;
#pragma unroll
  for (int i = 0; i < 32; i += 8)
    tile[ty + i][tx] = W[(size_t)(k0 + ty + i) * N + n0 + tx];
  __syncthreads();
#pragma unroll
  for (int i = 0; i < 32; i += 8)
    Wt[(size_t)(n0 + ty + i) * K + k0 + tx] = (__bf16)tile[tx][ty + i];
}

// ---------------------------------------------------------------------------
// bf16 GEMM: C[M][N] (f32) = A[M][K] * Bt[N][K]^T.
// Block: 256 thr = 8 waves, tile 128(M) x 128(N), BK=32, wave tile 32x64.
// Double-buffered async global->LDS staging (copies overlap WMMA).
// ---------------------------------------------------------------------------
#define BM 128
#define BN 128
#define BK 32
#define KSTR (BK + 8)   // 40 elems = 80B rows, 16B-aligned chunks

__global__ __launch_bounds__(256) void gemm_bf16_nt_kernel(
    const __bf16* __restrict__ A, const __bf16* __restrict__ Bt,
    float* __restrict__ C, int M, int N, int K) {
  __shared__ __align__(16) __bf16 sA[2][BM][KSTR];
  __shared__ __align__(16) __bf16 sB[2][BN][KSTR];
  const int bm   = blockIdx.y * BM;
  const int bn   = blockIdx.x * BN;
  const int tid  = threadIdx.x;
  const int wave = tid >> 5;
  const int wm   = (wave >> 1) * 32;   // 0,32,64,96
  const int wn   = (wave & 1) * 64;    // 0,64

  // each thread stages 16 elems (32B) of A and of B: row r0, cols c0..c0+15
  const int r0 = tid >> 1;         // 0..127
  const int c0 = (tid & 1) * 16;   // 0 or 16

  const unsigned ldsA0 = lds_addr32(&sA[0][r0][c0]);
  const unsigned ldsA1 = lds_addr32(&sA[1][r0][c0]);
  const unsigned ldsB0 = lds_addr32(&sB[0][r0][c0]);
  const unsigned ldsB1 = lds_addr32(&sB[1][r0][c0]);

  const __bf16* gA = &A [(size_t)(bm + r0) * K + c0];
  const __bf16* gB = &Bt[(size_t)(bn + r0) * K + c0];

  v8f acc[2][4] = {};

  // prologue: fill buffer 0
  async_copy_32B(ldsA0, gA);
  async_copy_32B(ldsB0, gB);
  async_wait0();
  __syncthreads();

  int cur = 0;
  for (int k0 = 0; k0 < K; k0 += BK) {
    // issue async copies for the next step into the other buffer
    if (k0 + BK < K) {
      async_copy_32B(cur ? ldsA0 : ldsA1, gA + k0 + BK);
      async_copy_32B(cur ? ldsB0 : ldsB1, gB + k0 + BK);
    }

    // compute from the current buffer
    const __bf16(*cA)[KSTR] = sA[cur];
    const __bf16(*cB)[KSTR] = sB[cur];
    v16bf a[2], b[4];
#pragma unroll
    for (int mt = 0; mt < 2; ++mt) a[mt] = frag_ld_a(&cA[wm + mt * 16][0], KSTR);
#pragma unroll
    for (int nt = 0; nt < 4; ++nt) b[nt] = frag_ld_b(&cB[wn + nt * 16][0], KSTR);

#pragma unroll
    for (int mt = 0; mt < 2; ++mt)
#pragma unroll
      for (int nt = 0; nt < 4; ++nt)
        acc[mt][nt] = WMMA_BF16(a[mt], b[nt], acc[mt][nt]);

    async_wait0();     // next buffer's copies have landed
    __syncthreads();   // everyone done reading cur + done filling next
    cur ^= 1;
  }

  const int lane = tid & 31;
  const int cn = lane & 15;
  const int cm = (lane >> 4) * 8;
#pragma unroll
  for (int mt = 0; mt < 2; ++mt)
#pragma unroll
    for (int nt = 0; nt < 4; ++nt)
#pragma unroll
      for (int j = 0; j < 8; ++j)
        C[(size_t)(bm + wm + mt * 16 + cm + j) * N + (bn + wn + nt * 16 + cn)] = acc[mt][nt][j];
}

// ---------------------------------------------------------------------------
// RMSNorm (D=128) + RoPE; src [B][S][Hn][128] f32 -> dst [B][Hn][Ltot][128] bf16
// One wave per head-vector; 4 elems/lane; pair (d, d+64) exchanged via shfl_xor(16).
// ---------------------------------------------------------------------------
__global__ __launch_bounds__(256) void qk_norm_rope_kernel(
    const float* __restrict__ src, const float* __restrict__ g,
    const float* __restrict__ cache, __bf16* __restrict__ dst,
    int S, int Hn, int Ltot, int posOff) {
  const int vec  = blockIdx.x * 8 + (threadIdx.x >> 5);
  const int lane = threadIdx.x & 31;
  const int h = vec % Hn;
  const int t = vec / Hn;
  const int s = t % S;
  const int b = t / S;

  const float4 x  = ((const float4*)(src + (size_t)vec * 128))[lane];
  float ss = x.x * x.x + x.y * x.y + x.z * x.z + x.w * x.w;
#pragma unroll
  for (int off = 16; off >= 1; off >>= 1) ss += __shfl_xor(ss, off, 32);
  const float inv = rsqrtf(ss * (1.0f / 128.0f) + 1e-6f);

  const float4 gg = ((const float4*)g)[lane];
  float4 xn;
  xn.x = x.x * inv * gg.x;
  xn.y = x.y * inv * gg.y;
  xn.z = x.z * inv * gg.z;
  xn.w = x.w * inv * gg.w;

  float4 other;
  other.x = __shfl_xor(xn.x, 16, 32);
  other.y = __shfl_xor(xn.y, 16, 32);
  other.z = __shfl_xor(xn.z, 16, 32);
  other.w = __shfl_xor(xn.w, 16, 32);

  const int fj = lane & 15;  // frequency-quad index
  const float4 cs = ((const float4*)(cache + (size_t)s * 128))[fj];
  const float4 sn = ((const float4*)(cache + (size_t)s * 128 + 64))[fj];

  float4 r;
  if (lane < 16) {  // x1*cos - x2*sin
    r.x = xn.x * cs.x - other.x * sn.x;
    r.y = xn.y * cs.y - other.y * sn.y;
    r.z = xn.z * cs.z - other.z * sn.z;
    r.w = xn.w * cs.w - other.w * sn.w;
  } else {          // x2*cos + x1*sin
    r.x = xn.x * cs.x + other.x * sn.x;
    r.y = xn.y * cs.y + other.y * sn.y;
    r.z = xn.z * cs.z + other.z * sn.z;
    r.w = xn.w * cs.w + other.w * sn.w;
  }

  __bf16* o = dst + (((size_t)b * Hn + h) * Ltot + posOff + s) * 128 + lane * 4;
  o[0] = (__bf16)r.x; o[1] = (__bf16)r.y; o[2] = (__bf16)r.z; o[3] = (__bf16)r.w;
}

// V pack: src [B][S][Hn][128] f32 -> dst [B][Hn][128][Ltot] bf16 (transposed)
__global__ __launch_bounds__(256) void v_pack_kernel(
    const float* __restrict__ src, __bf16* __restrict__ dst,
    int S, int Hn, int Ltot, int posOff, size_t n) {
  size_t i = (size_t)blockIdx.x * 256 + threadIdx.x;
  if (i >= n) return;
  const int d = (int)(i & 127);
  size_t t = i >> 7;
  const int h = (int)(t % Hn); t /= Hn;
  const int s = (int)(t % S);
  const int b = (int)(t / S);
  dst[(((size_t)b * Hn + h) * 128 + d) * Ltot + posOff + s] = (__bf16)src[i];
}

// ---------------------------------------------------------------------------
// Flash attention: 8 waves/block, one 16-query tile per wave, all waves share
// (batch, head). K and V chunks are staged ONCE per block into LDS with
// double-buffered async copies (8x less L2 traffic than per-wave loads).
//   Q [B][Hq][Si][128] bf16, K [B][Hkv][L][128] bf16, Vt [B][Hkv][128][L] bf16
//   Out [B][Si][Hq][128] bf16 (token-major for the out-projection GEMM)
// ---------------------------------------------------------------------------
#define PSTR 40   // probs row stride: 32 + 8 pad (80B, 16B-aligned)
#define KPAD 136  // K-tile row stride: 128 + 8 pad (272B, 16B-aligned, bank-skewed)
#define VPAD 40   // V-tile row stride: 32 + 8 pad

__global__ __launch_bounds__(256) void flash_attn_kernel(
    const __bf16* __restrict__ Q, const __bf16* __restrict__ Kb,
    const __bf16* __restrict__ Vt, __bf16* __restrict__ Out,
    int Si, int L, int Hq, int Hkv) {
  __shared__ __align__(16) __bf16 sK[2][32][KPAD];   // [key][d]
  __shared__ __align__(16) __bf16 sV[2][128][VPAD];  // [d][key]
  __shared__ __align__(16) __bf16 sP[8][16][PSTR];   // per-wave probs staging
  const int tid  = threadIdx.x;
  const int w    = tid >> 5;
  const int lane = tid & 31;
  const int qt   = blockIdx.x * 8 + w;
  const int h    = blockIdx.y;
  const int b    = blockIdx.z;
  const int hk   = h / (Hq / Hkv);
  const int q0   = qt * 16;

  const __bf16* Qt = Q  + (((size_t)b * Hq  + h ) * Si + q0) * 128;
  const __bf16* Kh = Kb + ((size_t)b * Hkv + hk) * (size_t)L * 128;
  const __bf16* Vh = Vt + ((size_t)b * Hkv + hk) * 128 * (size_t)L;

  // staging assignment: K tile 32x128 (8 chunks/row), V tile 128x32 (2 chunks/row)
  const int kr = tid >> 3;          // 0..31 key row
  const int kc = (tid & 7) * 16;    // 0..112
  const int vr = tid >> 1;          // 0..127 d row
  const int vc = (tid & 1) * 16;    // 0 or 16
  const unsigned ldsK0 = lds_addr32(&sK[0][kr][kc]);
  const unsigned ldsK1 = lds_addr32(&sK[1][kr][kc]);
  const unsigned ldsV0 = lds_addr32(&sV[0][vr][vc]);
  const unsigned ldsV1 = lds_addr32(&sV[1][vr][vc]);
  const __bf16* gK = Kh + (size_t)kr * 128 + kc;   // + kb*128
  const __bf16* gV = Vh + (size_t)vr * L + vc;     // + kb

  v16bf aq[4];
#pragma unroll
  for (int kc4 = 0; kc4 < 4; ++kc4) aq[kc4] = frag_ld_a(Qt + kc4 * 32, 128);

  v8f o[8] = {};
  float mrow[8], lrow[8];
#pragma unroll
  for (int j = 0; j < 8; ++j) { mrow[j] = -3.0e38f; lrow[j] = 0.0f; }

  const float scale = 0.08838834764831845f;  // 1/sqrt(128)
  const int cm   = (lane >> 4) * 8;
  const int cn   = lane & 15;
  const int arow = lane & 15;

  // prologue: stage chunk 0 into buffer 0
  async_copy_32B(ldsK0, gK);
  async_copy_32B(ldsV0, gV);
  async_wait0();
  __syncthreads();

  int cur = 0;
  for (int kb = 0; kb < L; kb += 32) {
    // issue async staging of the next key chunk into the other buffer
    if (kb + 32 < L) {
      async_copy_32B(cur ? ldsK0 : ldsK1, gK + (size_t)(kb + 32) * 128);
      async_copy_32B(cur ? ldsV0 : ldsV1, gV + (kb + 32));
    }
    const __bf16(*cK)[KPAD] = sK[cur];
    const __bf16(*cV)[VPAD] = sV[cur];

    // ---- scores: 16 queries x 32 keys, K-dim = 128 in 4 WMMA steps each ----
    v8f s0 = {}, s1 = {};
#pragma unroll
    for (int kc4 = 0; kc4 < 4; ++kc4) {
      v16bf k0f = frag_ld_b(&cK[0][kc4 * 32],  KPAD);
      v16bf k1f = frag_ld_b(&cK[16][kc4 * 32], KPAD);
      s0 = WMMA_BF16(aq[kc4], k0f, s0);
      s1 = WMMA_BF16(aq[kc4], k1f, s1);
    }

    // ---- online softmax (row m = cm+j lives on this lane/vgpr) ----
    float p0[8], p1[8];
#pragma unroll
    for (int j = 0; j < 8; ++j) {
      const float a0 = s0[j] * scale;
      const float a1 = s1[j] * scale;
      float cmax = fmaxf(a0, a1);
#pragma unroll
      for (int off = 8; off >= 1; off >>= 1) cmax = fmaxf(cmax, __shfl_xor(cmax, off, 32));
      const float nm    = fmaxf(mrow[j], cmax);
      const float alpha = __expf(mrow[j] - nm);
      p0[j] = __expf(a0 - nm);
      p1[j] = __expf(a1 - nm);
      float rs = p0[j] + p1[j];
#pragma unroll
      for (int off = 8; off >= 1; off >>= 1) rs += __shfl_xor(rs, off, 32);
      lrow[j] = lrow[j] * alpha + rs;
      mrow[j] = nm;
#pragma unroll
      for (int tt = 0; tt < 8; ++tt) o[tt][j] *= alpha;
    }

    // ---- stage probs through LDS: C-layout -> A-fragment layout ----
#pragma unroll
    for (int j = 0; j < 8; ++j) {
      sP[w][cm + j][cn]      = (__bf16)p0[j];
      sP[w][cm + j][16 + cn] = (__bf16)p1[j];
    }
    asm volatile("s_wait_dscnt 0" ::: "memory");  // same-wave DS RAW fence
    const v16bf ap = frag_ld_a(&sP[w][arow][0], PSTR);

    // ---- P*V accumulation: 8 N-tiles of 16 output dims ----
#pragma unroll
    for (int tt = 0; tt < 8; ++tt) {
      v16bf bv = frag_ld_b(&cV[tt * 16][0], VPAD);
      o[tt] = WMMA_BF16(ap, bv, o[tt]);
    }

    async_wait0();     // next chunk staged
    __syncthreads();   // all waves done reading cur
    cur ^= 1;
  }

  // ---- finalize and store [b][q][h][d] bf16 ----
#pragma unroll
  for (int tt = 0; tt < 8; ++tt) {
#pragma unroll
    for (int j = 0; j < 8; ++j) {
      const float val = o[tt][j] / lrow[j];
      Out[(((size_t)b * Si + q0 + cm + j) * Hq + h) * 128 + tt * 16 + cn] = (__bf16)val;
    }
  }
}

// ---------------------------------------------------------------------------
// Host launch
// ---------------------------------------------------------------------------
extern "C" void kernel_launch(void* const* d_in, const int* in_sizes, int n_in,
                              void* d_out, int out_size, void* d_ws, size_t ws_size,
                              hipStream_t stream) {
  (void)in_sizes; (void)n_in; (void)out_size; (void)ws_size;

  const float* hidden    = (const float*)d_in[0];
  const float* enc       = (const float*)d_in[1];
  const float* Wq        = (const float*)d_in[2];
  const float* Wk        = (const float*)d_in[3];
  const float* Wv        = (const float*)d_in[4];
  const float* Wak       = (const float*)d_in[5];
  const float* Wav       = (const float*)d_in[6];
  const float* Wo        = (const float*)d_in[7];
  const float* gq        = (const float*)d_in[8];
  const float* gk        = (const float*)d_in[9];
  const float* gak       = (const float*)d_in[10];
  const float* img_cache = (const float*)d_in[11];
  const float* txt_cache = (const float*)d_in[12];
  float* out = (float*)d_out;

  const int B = 2, Si = 2048, St = 512, H = 24, HKV = 8;
  const int dim   = H * 128;     // 3072
  const int kvdim = HKV * 128;   // 1024
  const int Mq = B * Si;         // 4096
  const int Mt = B * St;         // 1024
  const int L  = Si + St;        // 2560

  char* ws = (char*)d_ws;
  size_t off = 0;
  auto alloc = [&](size_t bytes) -> char* {
    char* p = ws + off;
    off += (bytes + 255) & ~(size_t)255;
    return p;
  };

  __bf16* hid_bf  = (__bf16*)alloc((size_t)Mq * dim * 2);
  __bf16* enc_bf  = (__bf16*)alloc((size_t)Mt * dim * 2);
  __bf16* WqT     = (__bf16*)alloc((size_t)dim * dim * 2);
  __bf16* WkT     = (__bf16*)alloc((size_t)kvdim * dim * 2);
  __bf16* WvT     = (__bf16*)alloc((size_t)kvdim * dim * 2);
  __bf16* WakT    = (__bf16*)alloc((size_t)kvdim * dim * 2);
  __bf16* WavT    = (__bf16*)alloc((size_t)kvdim * dim * 2);
  __bf16* WoT     = (__bf16*)alloc((size_t)dim * dim * 2);
  __bf16* Qbf     = (__bf16*)alloc((size_t)B * H * Si * 128 * 2);
  __bf16* Kbf     = (__bf16*)alloc((size_t)B * HKV * L * 128 * 2);
  __bf16* Vtb     = (__bf16*)alloc((size_t)B * HKV * L * 128 * 2);
  __bf16* attn_bf = (__bf16*)alloc((size_t)Mq * dim * 2);
  float*  qf      = (float*)alloc((size_t)Mq * dim * 4);
  float*  kf      = (float*)alloc((size_t)Mq * kvdim * 4);
  float*  vf      = (float*)alloc((size_t)Mq * kvdim * 4);
  float*  tkf     = (float*)alloc((size_t)Mt * kvdim * 4);
  float*  tvf     = (float*)alloc((size_t)Mt * kvdim * 4);

  // 1. activations f32 -> bf16
  {
    size_t n = (size_t)Mq * dim;
    cvt_f32_bf16_kernel<<<(unsigned)((n + 255) / 256), 256, 0, stream>>>(hidden, hid_bf, n);
    n = (size_t)Mt * dim;
    cvt_f32_bf16_kernel<<<(unsigned)((n + 255) / 256), 256, 0, stream>>>(enc, enc_bf, n);
  }
  // 2. weights: convert + transpose -> bf16 [N][K]
  transpose_cvt_kernel<<<dim3(dim / 32,   dim / 32), 256, 0, stream>>>(Wq,  WqT,  dim, dim);
  transpose_cvt_kernel<<<dim3(kvdim / 32, dim / 32), 256, 0, stream>>>(Wk,  WkT,  dim, kvdim);
  transpose_cvt_kernel<<<dim3(kvdim / 32, dim / 32), 256, 0, stream>>>(Wv,  WvT,  dim, kvdim);
  transpose_cvt_kernel<<<dim3(kvdim / 32, dim / 32), 256, 0, stream>>>(Wak, WakT, dim, kvdim);
  transpose_cvt_kernel<<<dim3(kvdim / 32, dim / 32), 256, 0, stream>>>(Wav, WavT, dim, kvdim);
  transpose_cvt_kernel<<<dim3(dim / 32,   dim / 32), 256, 0, stream>>>(Wo,  WoT,  dim, dim);

  // 3. projection GEMMs (WMMA bf16, f32 out)
  gemm_bf16_nt_kernel<<<dim3(dim / BN,   Mq / BM), 256, 0, stream>>>(hid_bf, WqT,  qf,  Mq, dim,   dim);
  gemm_bf16_nt_kernel<<<dim3(kvdim / BN, Mq / BM), 256, 0, stream>>>(hid_bf, WkT,  kf,  Mq, kvdim, dim);
  gemm_bf16_nt_kernel<<<dim3(kvdim / BN, Mq / BM), 256, 0, stream>>>(hid_bf, WvT,  vf,  Mq, kvdim, dim);
  gemm_bf16_nt_kernel<<<dim3(kvdim / BN, Mt / BM), 256, 0, stream>>>(enc_bf, WakT, tkf, Mt, kvdim, dim);
  gemm_bf16_nt_kernel<<<dim3(kvdim / BN, Mt / BM), 256, 0, stream>>>(enc_bf, WavT, tvf, Mt, kvdim, dim);

  // 4. RMSNorm + RoPE -> bf16 attention layouts
  qk_norm_rope_kernel<<<(Mq * H)   / 8, 256, 0, stream>>>(qf,  gq,  img_cache, Qbf, Si, H,   Si, 0);
  qk_norm_rope_kernel<<<(Mq * HKV) / 8, 256, 0, stream>>>(kf,  gk,  img_cache, Kbf, Si, HKV, L,  0);
  qk_norm_rope_kernel<<<(Mt * HKV) / 8, 256, 0, stream>>>(tkf, gak, txt_cache, Kbf, St, HKV, L,  Si);

  // 5. V -> transposed bf16 [b][hkv][d][key]
  {
    size_t n = (size_t)Mq * kvdim;
    v_pack_kernel<<<(unsigned)((n + 255) / 256), 256, 0, stream>>>(vf,  Vtb, Si, HKV, L, 0,  n);
    n = (size_t)Mt * kvdim;
    v_pack_kernel<<<(unsigned)((n + 255) / 256), 256, 0, stream>>>(tvf, Vtb, St, HKV, L, Si, n);
  }

  // 6. flash attention (16 q-tiles/wave-block x 24 heads x 2 batches)
  flash_attn_kernel<<<dim3(Si / 16 / 8, H, B), 256, 0, stream>>>(Qbf, Kbf, Vtb, attn_bf, Si, L, H, HKV);

  // 7. out projection -> f32 d_out
  gemm_bf16_nt_kernel<<<dim3(dim / BN, Mq / BM), 256, 0, stream>>>(attn_bf, WoT, out, Mq, dim, dim);
}
